// SimpleLSTM_52046413693429
// MI455X (gfx1250) — matrix operand order repaired and checked by reading, not compiled, not run on hardware
//
#include <hip/hip_runtime.h>

// CDNA5 (gfx1250) fused 2-layer LSTM + FC head, v2.
// - 128 workgroups x 128 threads (4 wave32); each WG owns a 16-row batch tile
//   for the whole T=512 sequence (both layers + FC fused; no [T,B,H] HBM
//   intermediates -> HBM traffic is just x (42 MB) + out).
// - All 28 weight B-fragments are register-resident across the time loop
//   (loaded once from LDS-swizzled f16 weights), so the steady-state step is
//   pure v_wmma + TRANS-pipe cell math + a small LDS h-exchange.
// - Branch-free sigmoid/tanh via v_exp_f32/v_rcp_f32 (co-executes with WMMA).
// - Ping-pong h staging: 2 workgroup barriers per step instead of 3.

typedef __attribute__((ext_vector_type(16))) _Float16     v16h;
typedef __attribute__((ext_vector_type(8)))  float        v8f;
typedef __attribute__((ext_vector_type(4)))  unsigned int v4u;

namespace {
constexpr int kB = 2048;   // batch
constexpr int kT = 512;    // time steps
constexpr int kI = 10;     // input features
constexpr int kH = 50;     // hidden size
constexpr int HP = 64;     // hidden padded to WMMA tiles
constexpr int ROWS = 16;   // batch rows per workgroup (= WMMA M)
constexpr int THREADS = 128;
constexpr int XROW = 32;   // padded x row length (halves) = one K chunk

// LDS weight region: blocks of 512 halves, block layout [n(16)][k(32)] col-major
constexpr int WIH0_OFF = 0;                   // 16 blocks (16 n-tiles x 1 K-chunk)
constexpr int WHH0_OFF = WIH0_OFF + 16 * 512; // 32 blocks (16 x 2)
constexpr int WIH1_OFF = WHH0_OFF + 32 * 512; // 32 blocks
constexpr int WHH1_OFF = WIH1_OFF + 32 * 512; // 32 blocks
constexpr int WTOT     = WHH1_OFF + 32 * 512; // 57344 halves = 112 KB
}

__device__ __forceinline__ v16h ld_pair(const _Float16* p0, const _Float16* p1) {
  union { v4u q[2]; v16h v; } u;
  u.q[0] = *reinterpret_cast<const v4u*>(p0);   // ds_load_b128
  u.q[1] = *reinterpret_cast<const v4u*>(p1);   // ds_load_b128
  return u.v;
}

__device__ __forceinline__ v8f wmma_f16(v16h a, v16h b, v8f c) {
  return __builtin_amdgcn_wmma_f32_16x16x32_f16(false, a, false, b, (short)0, c,
                                                false, false);
}

// Branch-free fast transcendentals on the TRANS pipe (v_exp_f32 / v_rcp_f32).
__device__ __forceinline__ float fast_sig(float x) {
  return __builtin_amdgcn_rcpf(1.0f + __builtin_amdgcn_exp2f(x * -1.442695041f));
}
__device__ __forceinline__ float fast_tanh(float x) {
  return 2.0f * __builtin_amdgcn_rcpf(1.0f + __builtin_amdgcn_exp2f(x * -2.885390082f)) - 1.0f;
}

__global__ __launch_bounds__(THREADS, 1)
void lstm2_fused_kernel(const float* __restrict__ x,
                        const float* __restrict__ w_ih0, const float* __restrict__ w_hh0,
                        const float* __restrict__ b_ih0, const float* __restrict__ b_hh0,
                        const float* __restrict__ w_ih1, const float* __restrict__ w_hh1,
                        const float* __restrict__ b_ih1, const float* __restrict__ b_hh1,
                        const float* __restrict__ w_fc,  const float* __restrict__ b_fc,
                        float* __restrict__ out)
{
  __shared__ alignas(16) _Float16 ldsW[WTOT];            // swizzled f16 weights (init only)
  __shared__ alignas(16) _Float16 xbuf[ROWS * XROW];     // x_t tile (A-frag source)
  __shared__ alignas(16) _Float16 h0buf[2][ROWS * HP];   // layer0 h, ping-pong
  __shared__ alignas(16) _Float16 h1buf[2][ROWS * HP];   // layer1 h, ping-pong
  __shared__ alignas(16) float    hfin[ROWS * HP];       // final-step h1 (f32) for FC

  const int tid  = threadIdx.x;
  const int lane = tid & 31;
  const int wv   = tid >> 5;              // wave id: owns hidden cols [wv*16, wv*16+16)
  const int rowBase = blockIdx.x * ROWS;

  // ---- one-time: convert fp32 weights -> padded f16 B-fragment layout in LDS ----
  for (int idx = tid; idx < WTOT; idx += THREADS) {
    const float* src; int Kreal, nch, rem;
    if (idx < WHH0_OFF)      { rem = idx - WIH0_OFF; src = w_ih0; Kreal = kI; nch = 1; }
    else if (idx < WIH1_OFF) { rem = idx - WHH0_OFF; src = w_hh0; Kreal = kH; nch = 2; }
    else if (idx < WHH1_OFF) { rem = idx - WIH1_OFF; src = w_ih1; Kreal = kH; nch = 2; }
    else                     { rem = idx - WHH1_OFF; src = w_hh1; Kreal = kH; nch = 2; }
    const int blk = rem >> 9, o = rem & 511, n = o >> 5, k = o & 31;
    const int nt = blk / nch, ch = blk % nch;
    const int col = nt * 16 + n;          // padded gate column 0..255
    const int g = col >> 6, hcol = col & 63, kin = ch * 32 + k;
    float v = 0.0f;
    if (hcol < kH && kin < Kreal) v = src[(g * kH + hcol) * Kreal + kin];
    ldsW[idx] = (_Float16)v;              // zero padding keeps padded h/c exactly 0
  }
  for (int i2 = tid; i2 < ROWS * XROW; i2 += THREADS) xbuf[i2] = (_Float16)0.0f;
  for (int i2 = tid; i2 < ROWS * HP; i2 += THREADS) {
    h0buf[0][i2] = (_Float16)0.0f; h0buf[1][i2] = (_Float16)0.0f;
    h1buf[0][i2] = (_Float16)0.0f; h1buf[1][i2] = (_Float16)0.0f;
  }
  __syncthreads();

  const int ncol = lane & 15;             // column within this wave's 16-wide block
  const int kh   = lane >> 4;             // K-half selector for A/B fragments
  const int hcolL = wv * 16 + ncol;
  const int mb = (lane < 16) ? 0 : 8;     // D-layout: VGPR r = row r (+8 for hi lanes)

  // ---- hoist all 28 weight B-fragments into registers (loop-invariant) ----
  const int bo = ncol * 32 + kh * 16;     // per-lane offset inside a 512-half block
  v16h wx[4], wh0a[4], wh0b[4], wi1a[4], wi1b[4], wh1a[4], wh1b[4];
  #pragma unroll
  for (int g = 0; g < 4; ++g) {
    const int nt = g * 4 + wv;
    const _Float16* p;
    p = &ldsW[WIH0_OFF + nt * 512 + bo];            wx[g]   = ld_pair(p, p + 8);
    p = &ldsW[WHH0_OFF + (nt * 2 + 0) * 512 + bo];  wh0a[g] = ld_pair(p, p + 8);
    p = &ldsW[WHH0_OFF + (nt * 2 + 1) * 512 + bo];  wh0b[g] = ld_pair(p, p + 8);
    p = &ldsW[WIH1_OFF + (nt * 2 + 0) * 512 + bo];  wi1a[g] = ld_pair(p, p + 8);
    p = &ldsW[WIH1_OFF + (nt * 2 + 1) * 512 + bo];  wi1b[g] = ld_pair(p, p + 8);
    p = &ldsW[WHH1_OFF + (nt * 2 + 0) * 512 + bo];  wh1a[g] = ld_pair(p, p + 8);
    p = &ldsW[WHH1_OFF + (nt * 2 + 1) * 512 + bo];  wh1b[g] = ld_pair(p, p + 8);
  }

  // ---- per-gate bias vectors, used directly as first-WMMA C operand ----
  v8f bias0v[4], bias1v[4];
  #pragma unroll
  for (int g = 0; g < 4; ++g) {
    const float b0 = (hcolL < kH) ? (b_ih0[g * kH + hcolL] + b_hh0[g * kH + hcolL]) : 0.0f;
    const float b1 = (hcolL < kH) ? (b_ih1[g * kH + hcolL] + b_hh1[g * kH + hcolL]) : 0.0f;
    #pragma unroll
    for (int e = 0; e < 8; ++e) { bias0v[g][e] = b0; bias1v[g][e] = b1; }
  }
  v8f c0, c1, h0v, h1v;
  #pragma unroll
  for (int e = 0; e < 8; ++e) { c0[e] = 0.0f; c1[e] = 0.0f; }

  int pp = 0;  // ping-pong index
  for (int t = 0; t < kT; ++t) {
    // stage x_t tile -> f16 LDS; prefetch next step's tile
    for (int e = tid; e < ROWS * kI; e += THREADS) {
      const int r = e / kI, i = e % kI;
      const size_t gidx = (size_t)(rowBase + r) * (kT * kI) + (size_t)t * kI + i;
      xbuf[r * XROW + i] = (_Float16)x[gidx];
      if (t + 1 < kT) __builtin_prefetch(&x[gidx + kI], 0, 0);
    }
    __syncthreads();  // B1: xbuf + previous-step h writes visible

    const _Float16* h0r = h0buf[pp];
    const _Float16* h1r = h1buf[pp];
    _Float16* h0w = h0buf[pp ^ 1];
    _Float16* h1w = h1buf[pp ^ 1];

    // A fragments (row = lane%16; this code's contiguous k order: lane/16 picks the K-half, same as B)
    v16h ax = ld_pair(&xbuf[ncol * XROW + kh * 16], &xbuf[ncol * XROW + kh * 16 + 8]);
    v16h ah0[2], ah1[2];
    #pragma unroll
    for (int ch = 0; ch < 2; ++ch) {
      const int o = ncol * HP + ch * 32 + kh * 16;
      ah0[ch] = ld_pair(&h0r[o], &h0r[o + 8]);
      ah1[ch] = ld_pair(&h1r[o], &h1r[o + 8]);
    }

    // ---- layer 0: gates = x@Wih0^T + h0@Whh0^T + b (weights in registers) ----
    v8f acc[4];
    #pragma unroll
    for (int g = 0; g < 4; ++g) {
      acc[g] = wmma_f16(ax,     wx[g],   bias0v[g]);
      acc[g] = wmma_f16(ah0[0], wh0a[g], acc[g]);
      acc[g] = wmma_f16(ah0[1], wh0b[g], acc[g]);
    }
    #pragma unroll
    for (int e = 0; e < 8; ++e) {  // LSTM cell, gate order i,f,g,o
      const float iv = fast_sig(acc[0][e]);
      const float fv = fast_sig(acc[1][e]);
      const float gv = fast_tanh(acc[2][e]);
      const float ov = fast_sig(acc[3][e]);
      const float cv = fv * c0[e] + iv * gv;
      c0[e] = cv;
      h0v[e] = ov * fast_tanh(cv);
    }
    #pragma unroll
    for (int r = 0; r < 8; ++r)
      h0w[(mb + r) * HP + wv * 16 + ncol] = (_Float16)h0v[r];
    __syncthreads();  // B2: new h0 visible to all waves

    v16h ah0n[2];
    #pragma unroll
    for (int ch = 0; ch < 2; ++ch) {
      const int o = ncol * HP + ch * 32 + kh * 16;
      ah0n[ch] = ld_pair(&h0w[o], &h0w[o + 8]);
    }

    // ---- layer 1: gates = h0_new@Wih1^T + h1@Whh1^T + b ----
    #pragma unroll
    for (int g = 0; g < 4; ++g) {
      acc[g] = wmma_f16(ah0n[0], wi1a[g], bias1v[g]);
      acc[g] = wmma_f16(ah0n[1], wi1b[g], acc[g]);
      acc[g] = wmma_f16(ah1[0],  wh1a[g], acc[g]);
      acc[g] = wmma_f16(ah1[1],  wh1b[g], acc[g]);
    }
    #pragma unroll
    for (int e = 0; e < 8; ++e) {
      const float iv = fast_sig(acc[0][e]);
      const float fv = fast_sig(acc[1][e]);
      const float gv = fast_tanh(acc[2][e]);
      const float ov = fast_sig(acc[3][e]);
      const float cv = fv * c1[e] + iv * gv;
      c1[e] = cv;
      h1v[e] = ov * fast_tanh(cv);
    }
    #pragma unroll
    for (int r = 0; r < 8; ++r)
      h1w[(mb + r) * HP + wv * 16 + ncol] = (_Float16)h1v[r];

    pp ^= 1;
  }

  // ---- FC head: out[b] = h1_last[b,:] . w_fc + b_fc (O == 1) ----
  #pragma unroll
  for (int r = 0; r < 8; ++r)
    hfin[(mb + r) * HP + wv * 16 + ncol] = h1v[r];  // f32, from live registers
  __syncthreads();
  if (tid < ROWS) {
    float s = b_fc[0];
    for (int c = 0; c < kH; ++c) s += hfin[tid * HP + c] * w_fc[c];
    out[rowBase + tid] = s;
  }
}

extern "C" void kernel_launch(void* const* d_in, const int* in_sizes, int n_in,
                              void* d_out, int out_size, void* d_ws, size_t ws_size,
                              hipStream_t stream) {
  (void)in_sizes; (void)n_in; (void)out_size; (void)d_ws; (void)ws_size;
  lstm2_fused_kernel<<<dim3(kB / ROWS), dim3(THREADS), 0, stream>>>(
      (const float*)d_in[0],
      (const float*)d_in[1], (const float*)d_in[2],
      (const float*)d_in[3], (const float*)d_in[4],
      (const float*)d_in[5], (const float*)d_in[6],
      (const float*)d_in[7], (const float*)d_in[8],
      (const float*)d_in[9], (const float*)d_in[10],
      (float*)d_out);
}
